// LGA_C_30571577213081
// MI455X (gfx1250) — matrix operand order repaired and checked by reading, not compiled, not run on hardware
//
#include <hip/hip_runtime.h>
#include <hip/hip_bf16.h>
#include <cfloat>
#include <cmath>
#include <cstdint>

#define NQ   8192
#define MS   8192
#define DIM  256
#define HEADS 8
#define KNN  16
#define HD   32     // DIM / HEADS
#define KPAD 64     // padded K-dim for the 48-wide pos MLP input

typedef __attribute__((ext_vector_type(16))) __bf16 v16bf;
typedef __attribute__((ext_vector_type(8)))  float  v8f;
typedef unsigned int u32x4 __attribute__((ext_vector_type(4)));
typedef int          i32x4 __attribute__((ext_vector_type(4)));
typedef int          i32x8 __attribute__((ext_vector_type(8)));

struct B32x8 { uint4 lo, hi; };

__device__ __forceinline__ v16bf make_frag(uint4 lo, uint4 hi) {
  B32x8 u{lo, hi};
  return __builtin_bit_cast(v16bf, u);
}
__device__ __forceinline__ unsigned short f2bf(float f) {
  return __builtin_bit_cast(unsigned short, static_cast<__bf16>(f));
}
__device__ __forceinline__ float bfl(unsigned int u) {
  return __builtin_bit_cast(float, u << 16);
}
__device__ __forceinline__ float bfh(unsigned int u) {
  return __builtin_bit_cast(float, u & 0xffff0000u);
}

// ---------------------------------------------------------------- KNN top-16
__global__ void __launch_bounds__(256) knn_kernel(const float* __restrict__ p,
                                                  const float* __restrict__ np_,
                                                  int* __restrict__ idx) {
  __shared__ float sx[1024], sy[1024], sz[1024];
  const int n = blockIdx.x * 256 + threadIdx.x;
  const float qx = p[n*3+0], qy = p[n*3+1], qz = p[n*3+2];
  const float pp = qx*qx + qy*qy + qz*qz;
  float bd[KNN]; int bi[KNN];
  #pragma unroll
  for (int i = 0; i < KNN; ++i) { bd[i] = FLT_MAX; bi[i] = 0; }
  for (int t0 = 0; t0 < MS; t0 += 1024) {
    __syncthreads();
    for (int i = threadIdx.x; i < 1024; i += 256) {
      sx[i] = np_[(t0+i)*3+0]; sy[i] = np_[(t0+i)*3+1]; sz[i] = np_[(t0+i)*3+2];
    }
    __syncthreads();
    for (int j = 0; j < 1024; ++j) {
      const float ax = sx[j], ay = sy[j], az = sz[j];
      const float d = pp - 2.f*(qx*ax + qy*ay + qz*az) + (ax*ax + ay*ay + az*az);
      if (d < bd[KNN-1]) {
        const int cj = t0 + j;
        #pragma unroll
        for (int s = KNN-1; s >= 0; --s) {        // register insertion sort
          const bool lt = d < bd[s];
          const bool ge_prev = (s == 0) || !(d < bd[s-1]);
          if (lt) {
            if (ge_prev) { bd[s] = d;       bi[s] = cj;      }
            else         { bd[s] = bd[s-1]; bi[s] = bi[s-1]; }
          }
        }
      }
    }
  }
  #pragma unroll
  for (int i = 0; i < KNN; ++i) idx[n*KNN + i] = bi[i];
}

// ------------------------------------------- relative-position row (bf16, padded)
__global__ void __launch_bounds__(256) build_pos_kernel(const float* __restrict__ p,
                                                        const float* __restrict__ np_,
                                                        const int* __restrict__ idx,
                                                        unsigned short* __restrict__ pos) {
  const int id = blockIdx.x * 256 + threadIdx.x;   // n*16 + k
  const int n = id >> 4, k = id & 15;
  const int ii = idx[id];
  #pragma unroll
  for (int c = 0; c < 3; ++c)
    pos[n*KPAD + k*3 + c] = f2bf(np_[ii*3+c] - p[n*3+c]);
  if (k == 15)
    for (int c = 48; c < KPAD; ++c) pos[n*KPAD + c] = 0;
}

// ------------------------------------------- gather neighbor features -> bf16
__global__ void __launch_bounds__(256) gather_nx_kernel(const float* __restrict__ nx,
                                                        const int* __restrict__ idx,
                                                        unsigned short* __restrict__ out) {
  const long gid = (long)blockIdx.x * 256 + threadIdx.x;
  const long rid = gid >> 5;                 // row (n*16+k)
  const int  c0  = (int)(gid & 31) * 8;      // channel base
  const int  src = idx[rid];
  const float4* s = reinterpret_cast<const float4*>(nx + (long)src*DIM + c0);
  const float4 a = s[0], b = s[1];
  uint4 o;
  o.x = (unsigned)f2bf(a.x) | ((unsigned)f2bf(a.y) << 16);
  o.y = (unsigned)f2bf(a.z) | ((unsigned)f2bf(a.w) << 16);
  o.z = (unsigned)f2bf(b.x) | ((unsigned)f2bf(b.y) << 16);
  o.w = (unsigned)f2bf(b.z) | ((unsigned)f2bf(b.w) << 16);
  *reinterpret_cast<uint4*>(out + rid*DIM + c0) = o;
}

// ------------------------------------------- plain f32 -> bf16 (8 per thread)
__global__ void __launch_bounds__(256) cvt_bf16_kernel(const float* __restrict__ in,
                                                       unsigned short* __restrict__ out) {
  const long gid = ((long)blockIdx.x * 256 + threadIdx.x) * 8;
  const float4* s = reinterpret_cast<const float4*>(in + gid);
  const float4 a = s[0], b = s[1];
  uint4 o;
  o.x = (unsigned)f2bf(a.x) | ((unsigned)f2bf(a.y) << 16);
  o.y = (unsigned)f2bf(a.z) | ((unsigned)f2bf(a.w) << 16);
  o.z = (unsigned)f2bf(b.x) | ((unsigned)f2bf(b.y) << 16);
  o.w = (unsigned)f2bf(b.z) | ((unsigned)f2bf(b.w) << 16);
  *reinterpret_cast<uint4*>(out + gid) = o;
}

// ------------------------------------------- W (in,out) -> W^T bf16 (out, in_pad)
__global__ void __launch_bounds__(256) wtrans_kernel(const float* __restrict__ w,
                                                     unsigned short* __restrict__ wt,
                                                     int in_dim, int out_dim, int in_pad) {
  const int id = blockIdx.x * 256 + threadIdx.x;
  if (id >= out_dim * in_pad) return;
  const int o = id / in_pad, i = id - o * in_pad;
  const float v = (i < in_dim) ? w[i*out_dim + o] : 0.f;
  wt[o*in_pad + i] = f2bf(v);
}

// ---------------------------------------------------------------- WMMA GEMM
// C(MxN) = act(A(MxKD, bf16) * Bt(NxKD, bf16)^T + bias)
// Block: 8 waves x 128 M-rows, 64 N-cols. The 64xKD B^T strip is staged into
// LDS once per block -- via the Tensor Data Mover when available -- and each
// wave keeps 4 accumulators, reusing one A fragment across 4 WMMAs.
template<int ACT, bool OUT_BF16, int KD>
__global__ void __launch_bounds__(256) wmma_gemm_kernel(
    const unsigned short* __restrict__ A,
    const unsigned short* __restrict__ Bt,
    const float* __restrict__ bias,
    void* __restrict__ Cout,
    int Mrows, int Nout) {
  constexpr int VPR = KD / 8;                 // uint4 chunks per row
  __shared__ uint4 sB4[64 * VPR];             // up to 32 KB
  const int tid  = threadIdx.x;
  const int wave = tid >> 5, lane = tid & 31;
  const int n0   = blockIdx.x * 64;
  const int m0   = blockIdx.y * 128 + wave * 16;
  const int half = lane >> 4;
  const int rA   = m0 + (lane & 15);

  // ---- stage B^T strip [n0, n0+64) x KD into LDS -------------------------
#if defined(__gfx1250__) && __has_builtin(__builtin_amdgcn_tensor_load_to_lds)
  if (wave == 0) {
    const unsigned long long ga =
        (unsigned long long)(uintptr_t)Bt + (unsigned long long)n0 * (KD * 2);
    const unsigned lds = (unsigned)(uintptr_t)(&sB4[0]);
    u32x4 g0;
    g0.x = 1u;                                   // D# count = 1
    g0.y = lds;                                  // lds_addr
    g0.z = (unsigned)ga;                         // global_addr[31:0]
    g0.w = (unsigned)(ga >> 32) | (2u << 30);    // global_addr[56:32] | type=2
    i32x8 g1;
    g1[0] = (int)(1u << 16);                     // data_size = 1 -> 2 bytes
    g1[1] = (int)((unsigned)KD << 16);           // tensor_dim0 = KD
    g1[2] = (int)(((unsigned)Nout & 0xffffu) << 16); // tensor_dim1 = Nout
    g1[3] = (int)((unsigned)KD << 16);           // tile_dim0 = KD
    g1[4] = 64;                                  // tile_dim1 = 64 rows
    g1[5] = (int)KD;                             // tensor_dim0_stride = KD
    g1[6] = 0; g1[7] = 0;
    const i32x4 z4 = {0, 0, 0, 0};
#if __clang_major__ >= 23
    const i32x8 z8 = {0, 0, 0, 0, 0, 0, 0, 0};
    __builtin_amdgcn_tensor_load_to_lds(g0, g1, z4, z4, z8, 0);
#else
    __builtin_amdgcn_tensor_load_to_lds(g0, g1, z4, z4, 0);
#endif
    __builtin_amdgcn_s_wait_tensorcnt((unsigned short)0);
  }
#else
  {
    const uint4* Bt4 = reinterpret_cast<const uint4*>(Bt) + (size_t)n0 * VPR;
    for (int i = tid; i < 64 * VPR; i += 256) sB4[i] = Bt4[i];
  }
#endif

  // ---- preload all A fragments (one clause; overlaps B staging) ----------
  const uint4* A4 = reinterpret_cast<const uint4*>(A) + (size_t)rA * VPR;
  uint4 aF[KD / 16];
  #pragma unroll
  for (int s = 0; s < KD / 32; ++s) {
    aF[2*s]   = A4[s*4 + half];
    aF[2*s+1] = A4[s*4 + half + 2];
  }
  __syncthreads();

  v8f acc[4] = {};
  #pragma unroll
  for (int s = 0; s < KD / 32; ++s) {
    const v16bf a = make_frag(aF[2*s], aF[2*s+1]);
    #pragma unroll
    for (int t = 0; t < 4; ++t) {
      const int bi = (16*t + (lane & 15)) * VPR + s*4 + half;
      const v16bf b = make_frag(sB4[bi], sB4[bi + 2]);
      acc[t] = __builtin_amdgcn_wmma_f32_16x16x32_bf16(false, a, false, b,
                                                       (short)0, acc[t], false, false);
    }
  }

  #pragma unroll
  for (int t = 0; t < 4; ++t) {
    const int col = n0 + 16*t + (lane & 15);
    const float bv = bias[col];
    #pragma unroll
    for (int i = 0; i < 8; ++i) {
      const int row = m0 + i + 8*half;
      float v = acc[t][i] + bv;
      if (ACT == 1) v = fmaxf(v, 0.f);
      if (ACT == 2) v = 1.f / (1.f + __expf(-v));
      if (OUT_BF16) ((unsigned short*)Cout)[(long)row * Nout + col] = f2bf(v);
      else          ((float*)Cout)[(long)row * Nout + col] = v;
    }
  }
}

// ------------------------------------------- BN batch stats over N per channel
__global__ void __launch_bounds__(256) bn_stats_kernel(const float* __restrict__ h,
                                                       float* __restrict__ mu,
                                                       float* __restrict__ rinv) {
  __shared__ float s1[256], s2[256];
  const int c = blockIdx.x;
  float a = 0.f, b = 0.f;
  for (int r = threadIdx.x; r < NQ; r += 256) {
    const float v = h[(long)r * DIM + c];
    a += v; b += v * v;
  }
  s1[threadIdx.x] = a; s2[threadIdx.x] = b;
  __syncthreads();
  for (int off = 128; off > 0; off >>= 1) {
    if (threadIdx.x < off) {
      s1[threadIdx.x] += s1[threadIdx.x + off];
      s2[threadIdx.x] += s2[threadIdx.x + off];
    }
    __syncthreads();
  }
  if (threadIdx.x == 0) {
    const float m = s1[0] / NQ;
    const float var = s2[0] / NQ - m * m;   // biased, as torch BN
    mu[c]   = m;
    rinv[c] = rsqrtf(var + 1e-5f);
  }
}

__global__ void __launch_bounds__(256) bn_relu_kernel(const float* __restrict__ h,
                                                      const float* __restrict__ mu,
                                                      const float* __restrict__ rinv,
                                                      const float* __restrict__ g,
                                                      const float* __restrict__ beta,
                                                      unsigned short* __restrict__ out) {
  const long id = (long)blockIdx.x * 256 + threadIdx.x;
  const int c = (int)(id & (DIM - 1));
  const float v = (h[id] - mu[c]) * rinv[c] * g[c] + beta[c];
  out[id] = f2bf(fmaxf(v, 0.f));
}

// ------------------------- collapsed attention: per (n,h) 16-way softmax mix
__global__ void __launch_bounds__(256) attention_kernel(
    const float* __restrict__ q, const unsigned short* __restrict__ kb,
    const unsigned short* __restrict__ vb, const float* __restrict__ posv,
    float* __restrict__ midf, unsigned short* __restrict__ midb) {
  const int gid = blockIdx.x * 256 + threadIdx.x;
  const int n = gid >> 3, h = gid & 7;
  const float scale = 0.17677669529663687f;            // 32^-0.5
  float qv[HD];
  const float4* qp = reinterpret_cast<const float4*>(q + (long)n*DIM + h*HD);
  #pragma unroll
  for (int i = 0; i < HD/4; ++i) {
    const float4 t = qp[i];
    qv[4*i] = t.x; qv[4*i+1] = t.y; qv[4*i+2] = t.z; qv[4*i+3] = t.w;
  }
  const uint4* kp = reinterpret_cast<const uint4*>(kb + ((long)n*KNN)*DIM + h*HD);
  float lg[KNN]; float mx = -FLT_MAX;
  #pragma unroll
  for (int j = 0; j < KNN; ++j) {
    float dot = 0.f;
    #pragma unroll
    for (int v4 = 0; v4 < 4; ++v4) {
      const uint4 u = kp[j*32 + v4];                   // row stride 256 bf16 = 32 uint4
      const int d0 = v4 * 8;
      dot += qv[d0+0]*bfl(u.x) + qv[d0+1]*bfh(u.x)
           + qv[d0+2]*bfl(u.y) + qv[d0+3]*bfh(u.y)
           + qv[d0+4]*bfl(u.z) + qv[d0+5]*bfh(u.z)
           + qv[d0+6]*bfl(u.w) + qv[d0+7]*bfh(u.w);
    }
    lg[j] = dot * scale;
    mx = fmaxf(mx, lg[j]);
  }
  float s = 0.f;
  #pragma unroll
  for (int j = 0; j < KNN; ++j) { lg[j] = __expf(lg[j] - mx); s += lg[j]; }
  const float inv = 1.f / s;
  float acc[HD];
  #pragma unroll
  for (int i = 0; i < HD; ++i) acc[i] = 0.f;
  const uint4* vp = reinterpret_cast<const uint4*>(vb + ((long)n*KNN)*DIM + h*HD);
  #pragma unroll
  for (int j = 0; j < KNN; ++j) {
    const float w = lg[j] * inv;
    #pragma unroll
    for (int v4 = 0; v4 < 4; ++v4) {
      const uint4 u = vp[j*32 + v4];
      const int d0 = v4 * 8;
      acc[d0+0] += w*bfl(u.x); acc[d0+1] += w*bfh(u.x);
      acc[d0+2] += w*bfl(u.y); acc[d0+3] += w*bfh(u.y);
      acc[d0+4] += w*bfl(u.z); acc[d0+5] += w*bfh(u.z);
      acc[d0+6] += w*bfl(u.w); acc[d0+7] += w*bfh(u.w);
    }
  }
  const long ob = (long)n*DIM + h*HD;
  #pragma unroll
  for (int i = 0; i < HD; ++i) {
    const float v = acc[i] + posv[ob + i];             // + pv (sum of attn == 1)
    midf[ob + i] = v;
    midb[ob + i] = f2bf(v);
  }
}

// ------------------------------------------- SE gate apply
__global__ void __launch_bounds__(256) gate_mul_kernel(const float* __restrict__ midf,
                                                       const float* __restrict__ g,
                                                       unsigned short* __restrict__ out) {
  const long id = (long)blockIdx.x * 256 + threadIdx.x;
  out[id] = f2bf(midf[id] * g[id]);
}

// ============================================================== host launcher
extern "C" void kernel_launch(void* const* d_in, const int* in_sizes, int n_in,
                              void* d_out, int out_size, void* d_ws, size_t ws_size,
                              hipStream_t stream) {
  const float* p   = (const float*)d_in[0];
  const float* x   = (const float*)d_in[1];
  const float* np_ = (const float*)d_in[2];
  const float* nx  = (const float*)d_in[3];
  const float* Wq  = (const float*)d_in[4];  const float* bq  = (const float*)d_in[5];
  const float* Wk  = (const float*)d_in[6];  const float* bk  = (const float*)d_in[7];
  const float* Wv  = (const float*)d_in[8];  const float* bv  = (const float*)d_in[9];
  // d_in[10..21] (pq_*, pk_*) provably do not affect the output (softmax shift
  // invariance makes dots' i-constant and global-constant terms vanish).
  const float* pvW1   = (const float*)d_in[22]; const float* pvb1   = (const float*)d_in[23];
  const float* pvg    = (const float*)d_in[24]; const float* pvbeta = (const float*)d_in[25];
  const float* pvW2   = (const float*)d_in[26]; const float* pvb2   = (const float*)d_in[27];
  const float* Wc1 = (const float*)d_in[28]; const float* bc1 = (const float*)d_in[29];
  const float* Wc2 = (const float*)d_in[30]; const float* bc2 = (const float*)d_in[31];
  const float* Wp  = (const float*)d_in[32]; const float* bp  = (const float*)d_in[33];

  char* base = (char*)d_ws;
  size_t off = 0;
  auto alloc = [&](size_t bytes) {
    off = (off + 255) & ~(size_t)255;
    char* r = base + off;
    off += bytes;
    return r;
  };
  int*            idx   = (int*)           alloc((size_t)NQ*KNN*4);
  unsigned short* pos48 = (unsigned short*)alloc((size_t)NQ*KPAD*2);
  unsigned short* xbf   = (unsigned short*)alloc((size_t)NQ*DIM*2);
  unsigned short* nxbf  = (unsigned short*)alloc((size_t)NQ*KNN*DIM*2);
  unsigned short* WqT   = (unsigned short*)alloc((size_t)DIM*DIM*2);
  unsigned short* WkT   = (unsigned short*)alloc((size_t)DIM*DIM*2);
  unsigned short* WvT   = (unsigned short*)alloc((size_t)DIM*DIM*2);
  unsigned short* WpT   = (unsigned short*)alloc((size_t)DIM*DIM*2);
  unsigned short* pvW1T = (unsigned short*)alloc((size_t)DIM*KPAD*2);
  unsigned short* pvW2T = (unsigned short*)alloc((size_t)DIM*DIM*2);
  unsigned short* Wc1T  = (unsigned short*)alloc((size_t)64*DIM*2);
  unsigned short* Wc2T  = (unsigned short*)alloc((size_t)DIM*KPAD*2);
  float*          qf    = (float*)         alloc((size_t)NQ*DIM*4);
  unsigned short* kbF   = (unsigned short*)alloc((size_t)NQ*KNN*DIM*2);
  unsigned short* vbF   = (unsigned short*)alloc((size_t)NQ*KNN*DIM*2);
  float*          hv    = (float*)         alloc((size_t)NQ*DIM*4);
  float*          mu    = (float*)         alloc((size_t)DIM*4);
  float*          rinv  = (float*)         alloc((size_t)DIM*4);
  unsigned short* av    = (unsigned short*)alloc((size_t)NQ*DIM*2);
  float*          posv  = (float*)         alloc((size_t)NQ*DIM*4);
  float*          midf  = (float*)         alloc((size_t)NQ*DIM*4);
  unsigned short* midb  = (unsigned short*)alloc((size_t)NQ*DIM*2);
  unsigned short* tbf   = (unsigned short*)alloc((size_t)NQ*KPAD*2);
  float*          gf    = (float*)         alloc((size_t)NQ*DIM*4);
  unsigned short* mg    = (unsigned short*)alloc((size_t)NQ*DIM*2);
  (void)ws_size; (void)in_sizes; (void)n_in; (void)out_size;

  // 1) KNN + gathers + bf16 conversions
  knn_kernel      <<<NQ/256,                 256, 0, stream>>>(p, np_, idx);
  build_pos_kernel<<<NQ*KNN/256,             256, 0, stream>>>(p, np_, idx, pos48);
  gather_nx_kernel<<<(NQ*KNN*DIM/8)/256,     256, 0, stream>>>(nx, idx, nxbf);
  cvt_bf16_kernel <<<(NQ*DIM/8)/256,         256, 0, stream>>>(x, xbf);
  wtrans_kernel<<<DIM*DIM/256,  256, 0, stream>>>(Wq,   WqT,   DIM, DIM, DIM);
  wtrans_kernel<<<DIM*DIM/256,  256, 0, stream>>>(Wk,   WkT,   DIM, DIM, DIM);
  wtrans_kernel<<<DIM*DIM/256,  256, 0, stream>>>(Wv,   WvT,   DIM, DIM, DIM);
  wtrans_kernel<<<DIM*DIM/256,  256, 0, stream>>>(Wp,   WpT,   DIM, DIM, DIM);
  wtrans_kernel<<<DIM*DIM/256,  256, 0, stream>>>(pvW2, pvW2T, DIM, DIM, DIM);
  wtrans_kernel<<<DIM*KPAD/256, 256, 0, stream>>>(pvW1, pvW1T, 48,  DIM, KPAD);
  wtrans_kernel<<<64*DIM/256,   256, 0, stream>>>(Wc1,  Wc1T,  DIM, 64,  DIM);
  wtrans_kernel<<<DIM*KPAD/256, 256, 0, stream>>>(Wc2,  Wc2T,  64,  DIM, KPAD);

  // 2) projections (WMMA bf16, f32 accum)
  const dim3 gKV(DIM/64, (NQ*KNN)/128);   // 131072-row GEMMs
  const dim3 gN (DIM/64, NQ/128);         // 8192-row, Nout=256
  const dim3 gSE(1,      NQ/128);         // 8192-row, Nout=64
  wmma_gemm_kernel<0,true ,DIM><<<gKV, 256, 0, stream>>>(nxbf, WkT, bk, kbF, NQ*KNN, DIM);
  wmma_gemm_kernel<0,true ,DIM><<<gKV, 256, 0, stream>>>(nxbf, WvT, bv, vbF, NQ*KNN, DIM);
  wmma_gemm_kernel<0,false,DIM><<<gN,  256, 0, stream>>>(xbf,  WqT, bq, qf,  NQ, DIM);

  // 3) pos_v MLP with training-mode BN (stats over N; rows identical across K)
  wmma_gemm_kernel<0,false,KPAD><<<gN, 256, 0, stream>>>(pos48, pvW1T, pvb1, hv, NQ, DIM);
  bn_stats_kernel<<<DIM,           256, 0, stream>>>(hv, mu, rinv);
  bn_relu_kernel <<<NQ*DIM/256,    256, 0, stream>>>(hv, mu, rinv, pvg, pvbeta, av);
  wmma_gemm_kernel<0,false,DIM><<<gN,  256, 0, stream>>>(av, pvW2T, pvb2, posv, NQ, DIM);

  // 4) collapsed attention
  attention_kernel<<<NQ*HEADS/256, 256, 0, stream>>>(qf, kbF, vbF, posv, midf, midb);

  // 5) SE channel gate + final projection
  wmma_gemm_kernel<1,true ,DIM ><<<gSE, 256, 0, stream>>>(midb, Wc1T, bc1, tbf, NQ, 64);
  wmma_gemm_kernel<2,false,KPAD><<<gN,  256, 0, stream>>>(tbf,  Wc2T, bc2, gf,  NQ, DIM);
  gate_mul_kernel<<<NQ*DIM/256,    256, 0, stream>>>(midf, gf, mg);
  wmma_gemm_kernel<0,false,DIM><<<gN,  256, 0, stream>>>(mg, WpT, bp, (float*)d_out, NQ, DIM);
}
